// PairwiseCosineMLP_20555713478745
// MI455X (gfx1250) — matrix operand order repaired and checked by reading, not compiled, not run on hardware
//
#include <hip/hip_runtime.h>

typedef __attribute__((ext_vector_type(16))) _Float16     v16h;
typedef __attribute__((ext_vector_type(8)))  float        v8f;
typedef __attribute__((ext_vector_type(4)))  unsigned int v4u;
typedef __attribute__((ext_vector_type(8)))  unsigned int v8u;

#define KV      16
#define DD      128
#define HID     128
#define NOUT    128
#define NPAIRS  120
#define WAVES   4
#define TILE_B  16                 // batches per wave
#define BLK_B   (WAVES * TILE_B)   // batches per block

// Dynamic LDS layout:
//  [0,      32768) : W1^T as f16, 128x128 (cols of W1 contiguous; src rows >=120 zeroed)
//  [32768,  65536) : W2^T as f16, 128x128
//  [65536 + w*25600): per-wave scratch:
//        +0     : sHf[2]  raw H tile f32 16x128, double buffered (2 x 8 KB)
//                 (buffer 0 reused at the end as the f32 out tile)
//        +16384 : sHn     f16 16x128 (4 KB)   (reused as hidden activations)
//        +20480 : sF      f16 16x128 (4 KB)   features
//        +24576 : sG      f32 16x16  (1 KB)   gram
#define PER_WAVE   25600
#define SMEM_BYTES (65536 + WAVES * PER_WAVE)

// upper-triangle (i<j) gram flat indices i*16+j in triu order; pads -> 0
// (pad feature values are don't-care: W1 rows 120..127 are zeroed in LDS)
__device__ __constant__ unsigned char PAIR_IDX[128] = {
      1,  2,  3,  4,  5,  6,  7,  8,  9, 10, 11, 12, 13, 14, 15,
     18, 19, 20, 21, 22, 23, 24, 25, 26, 27, 28, 29, 30, 31,
     35, 36, 37, 38, 39, 40, 41, 42, 43, 44, 45, 46, 47,
     52, 53, 54, 55, 56, 57, 58, 59, 60, 61, 62, 63,
     69, 70, 71, 72, 73, 74, 75, 76, 77, 78, 79,
     86, 87, 88, 89, 90, 91, 92, 93, 94, 95,
    103,104,105,106,107,108,109,110,111,
    120,121,122,123,124,125,126,127,
    137,138,139,140,141,142,143,
    154,155,156,157,158,159,
    171,172,173,174,175,
    188,189,190,191,
    205,206,207,
    222,223,
    239,
      0,  0,  0,  0,  0,  0,  0,  0
};

__device__ __forceinline__ v8f wmma_f16(v16h a, v16h b, v8f c) {
    return __builtin_amdgcn_wmma_f32_16x16x32_f16(
        false, a, false, b, (short)0, c, false, false);
}

// ---- Tensor Data Mover: 1-D (nelem x 1) f32 tile, global <-> LDS ----------
// D# per CDNA5 ISA: group0 = {count/flags, lds_addr,
// global_addr[31:0], global_addr[56:32] | type=2}; group1 = {data_size=4B,
// tensor_dim0=nelem, tensor_dim1=1, tile_dim0=nelem, tile_dim1=1, stride}.
__device__ __forceinline__ void tdm_desc(v4u& g0, v8u& g1,
                                         unsigned int lds_addr,
                                         const void* gaddr,
                                         unsigned int nelem) {
    unsigned long long ga = (unsigned long long)(size_t)gaddr;
    g0[0] = 1u;                                            // count=1, user load
    g0[1] = lds_addr;                                      // LDS byte address
    g0[2] = (unsigned int)ga;                              // global addr lo
    g0[3] = (unsigned int)((ga >> 32) & 0x1FFFFFFu) | (2u << 30); // hi | type=2
    g1[0] = (2u << 16);                                    // data_size = 4 bytes
    g1[1] = (nelem & 0xFFFFu) << 16;                       // tensor_dim0[15:0]
    g1[2] = ((nelem >> 16) & 0xFFFFu) | (1u << 16);        // dim0[31:16], dim1=1
    g1[3] = (nelem & 0xFFFFu) << 16;                       // tile_dim0
    g1[4] = 1u;                                            // tile_dim1 = 1
    g1[5] = nelem;                                         // dim0_stride lo32
    g1[6] = 0u;
    g1[7] = 0u;
}

__device__ __forceinline__ void tdm_load_1d(unsigned int lds_addr,
                                            const void* gaddr,
                                            unsigned int nelem) {
    v4u g0; v8u g1;
    tdm_desc(g0, g1, lds_addr, gaddr, nelem);
    asm volatile("tensor_load_to_lds %0, %1" :: "s"(g0), "s"(g1) : "memory");
}

__device__ __forceinline__ void tdm_store_1d(unsigned int lds_addr,
                                             void* gaddr,
                                             unsigned int nelem) {
    v4u g0; v8u g1;
    tdm_desc(g0, g1, lds_addr, gaddr, nelem);
    asm volatile("tensor_store_from_lds %0, %1" :: "s"(g0), "s"(g1) : "memory");
}

__global__ void PairwiseCosineMLP_20555713478745_kernel(
    const float* __restrict__ H,   // (B, 16, 128)
    const float* __restrict__ W1,  // (120, 128)
    const float* __restrict__ b1,  // (128,)
    const float* __restrict__ W2,  // (128, 128)
    const float* __restrict__ b2,  // (128,)
    float* __restrict__ out,       // (B, 128)
    int nB)
{
    extern __shared__ char smem[];
    _Float16* sW1t = (_Float16*)smem;             // [n*128 + k]
    _Float16* sW2t = (_Float16*)(smem + 32768);   // [n*128 + k]

    const int tid  = threadIdx.x;
    const int wid  = tid >> 5;
    const int lane = tid & 31;

    // ---- stage transposed f16 weights once per block (W1 zero-padded to k=128)
    for (int idx = tid; idx < HID * HID; idx += WAVES * 32) {
        const int k = idx >> 7, n = idx & 127;
        sW1t[n * HID + k] = (k < NPAIRS) ? (_Float16)W1[idx] : (_Float16)0.0f;
        sW2t[n * HID + k] = (_Float16)W2[idx];
    }
    __syncthreads();

    char*     wbase  = smem + 65536 + wid * PER_WAVE;
    float*    sHf    = (float*)wbase;                 // 2 x 16x128 f32
    _Float16* sHn    = (_Float16*)(wbase + 16384);    // 16x128 f16
    _Float16* sF     = (_Float16*)(wbase + 20480);    // 16x128 f16
    float*    sG     = (float*)(wbase + 24576);       // 16x16 f32
    const unsigned int ldsHf  = (unsigned int)(size_t)(void*)sHf;
    const unsigned int ldsOut = ldsHf;                // reuse buffer 0 for output

    const int bw = blockIdx.x * BLK_B + wid * TILE_B;
    if (bw >= nB) return;

    const int mrow = lane & 15;
    const int khi  = lane >> 4;

    // prime TDM double buffer with batch bw+0
    tdm_load_1d(ldsHf, H + (size_t)bw * (KV * DD), KV * DD);

    // ======== per-batch: normalize rows, Gram via WMMA, cos features ========
    for (int bt = 0; bt < TILE_B; ++bt) {
        if (bt + 1 < TILE_B) {
            tdm_load_1d(ldsHf + ((bt + 1) & 1) * 8192,
                        H + (size_t)(bw + bt + 1) * (KV * DD), KV * DD);
            __builtin_amdgcn_s_wait_tensorcnt(1);   // batch bt's tile landed
        } else {
            __builtin_amdgcn_s_wait_tensorcnt(0);
        }
        const float* hb = sHf + (bt & 1) * 2048;    // this batch's 16x128 f32

        // two lanes per row: lanes (2r, 2r+1) cover row r's two 64-float halves
        const int r    = lane >> 1;
        const int half = lane & 1;
        const float4* src4 = (const float4*)(hb + r * DD + half * 64);

        float ss = 0.0f;
        #pragma unroll
        for (int i = 0; i < 16; ++i) {
            float4 v = src4[i];
            ss += v.x * v.x + v.y * v.y + v.z * v.z + v.w * v.w;
        }
        ss += __shfl_xor(ss, 1);
        const float inv = 1.0f / fmaxf(sqrtf(ss), 1e-12f);

        _Float16* drow = sHn + r * DD + half * 64;
        #pragma unroll
        for (int i = 0; i < 16; ++i) {
            float4 v = src4[i];
            drow[i * 4 + 0] = (_Float16)(v.x * inv);
            drow[i * 4 + 1] = (_Float16)(v.y * inv);
            drow[i * 4 + 2] = (_Float16)(v.z * inv);
            drow[i * 4 + 3] = (_Float16)(v.w * inv);
        }

        // gram = Hn * Hn^T : K=128 in 4 chunks of 32
        v8f acc = {};
        #pragma unroll
        for (int kc = 0; kc < 4; ++kc) {
            const int kb = kc * 32;
            v16h a, bm;
            #pragma unroll
            for (int j = 0; j < 8; ++j) {
                a[j]     = sHn[mrow * DD + kb + khi * 16 + j];
                a[8 + j] = sHn[mrow * DD + kb + khi * 16 + 8 + j];
            }
            #pragma unroll
            for (int j = 0; j < 16; ++j)            // B = Hn^T: col n = row n of Hn
                bm[j] = sHn[mrow * DD + kb + khi * 16 + j];
            acc = wmma_f16(a, bm, acc);
        }
        #pragma unroll
        for (int v = 0; v < 8; ++v)
            sG[(v + khi * 8) * 16 + mrow] = acc[v];

        // branch-free feature gather via constant pair-index table
        #pragma unroll
        for (int q = 0; q < 4; ++q) {
            const int p = lane + q * 32;
            sF[bt * HID + p] = (_Float16)sG[PAIR_IDX[p]];
        }
    }

    // ======== hidden = relu(F @ W1 + b1), f16 into sHn (reuse) ========
    #pragma unroll
    for (int nt = 0; nt < 8; ++nt) {
        const int n = nt * 16 + mrow;
        v8f acc = {};
        #pragma unroll
        for (int kc = 0; kc < 4; ++kc) {
            const int kb = kc * 32;
            v16h a, bm;
            #pragma unroll
            for (int j = 0; j < 8; ++j) {
                a[j]     = sF[mrow * HID + kb + khi * 16 + j];
                a[8 + j] = sF[mrow * HID + kb + khi * 16 + 8 + j];
            }
            #pragma unroll
            for (int j = 0; j < 16; ++j)            // contiguous: W1^T column n
                bm[j] = sW1t[n * HID + kb + khi * 16 + j];
            acc = wmma_f16(a, bm, acc);
        }
        const float bb = b1[n];
        #pragma unroll
        for (int v = 0; v < 8; ++v)
            sHn[(v + khi * 8) * HID + n] = (_Float16)fmaxf(acc[v] + bb, 0.0f);
    }

    // ======== out = hidden @ W2 + b2, f32 tile into LDS, drained by TDM ========
    float* sOut = sHf;   // buffer 0 free after the bt loop
    #pragma unroll
    for (int nt = 0; nt < 8; ++nt) {
        const int n = nt * 16 + mrow;
        v8f acc = {};
        #pragma unroll
        for (int kc = 0; kc < 4; ++kc) {
            const int kb = kc * 32;
            v16h a, bm;
            #pragma unroll
            for (int j = 0; j < 8; ++j) {
                a[j]     = sHn[mrow * HID + kb + khi * 16 + j];
                a[8 + j] = sHn[mrow * HID + kb + khi * 16 + 8 + j];
            }
            #pragma unroll
            for (int j = 0; j < 16; ++j)
                bm[j] = sW2t[n * HID + kb + khi * 16 + j];
            acc = wmma_f16(a, bm, acc);
        }
        const float bb = b2[n];
        #pragma unroll
        for (int v = 0; v < 8; ++v)
            sOut[(v + khi * 8) * NOUT + n] = acc[v] + bb;
    }

    // make LDS writes visible to the TDM, then DMA the 8KB out-tile to global.
    // (16 consecutive batches -> out rows bw..bw+15 are contiguous in memory)
    asm volatile("s_wait_dscnt 0x0" ::: "memory");
    tdm_store_1d(ldsOut, out + (size_t)bw * NOUT, TILE_B * NOUT);
    // S_ENDPGM performs an implicit wait-idle, covering the outstanding TDM store.
}

extern "C" void kernel_launch(void* const* d_in, const int* in_sizes, int n_in,
                              void* d_out, int out_size, void* d_ws, size_t ws_size,
                              hipStream_t stream) {
    const float* H  = (const float*)d_in[0];
    const float* W1 = (const float*)d_in[1];
    const float* b1 = (const float*)d_in[2];
    const float* W2 = (const float*)d_in[3];
    const float* b2 = (const float*)d_in[4];
    float* out = (float*)d_out;

    const int nB = in_sizes[0] / (KV * DD);   // 65536 for the reference shapes

    (void)hipFuncSetAttribute(
        reinterpret_cast<const void*>(PairwiseCosineMLP_20555713478745_kernel),
        hipFuncAttributeMaxDynamicSharedMemorySize, SMEM_BYTES);

    dim3 grid((nB + BLK_B - 1) / BLK_B);
    dim3 block(WAVES * 32);
    PairwiseCosineMLP_20555713478745_kernel<<<grid, block, SMEM_BYTES, stream>>>(
        H, W1, b1, W2, b2, out, nB);
}
